// future_STGAT_decoder_41429254537487
// MI455X (gfx1250) — matrix operand order repaired and checked by reading, not compiled, not run on hardware
//
#include <hip/hip_runtime.h>
#include <hip/hip_bf16.h>

typedef __attribute__((ext_vector_type(16))) __bf16 v16bf;
typedef __attribute__((ext_vector_type(8)))  float  v8f;

#define Hd   512          // hidden size
#define Gd   2048         // 4*H gates
#define Kn   20           // samples
#define Np   4096         // pedestrians
#define ROWS (Kn*Np)      // 81920 flattened batch rows
#define FUT  12
#define OBS  8
#define BM   128          // rows per block (8 strips of 16)

__device__ __forceinline__ unsigned short f32_to_bf16(float f) {
    unsigned int u = __float_as_uint(f);
    u += 0x7FFFu + ((u >> 16) & 1u);          // round-to-nearest-even
    return (unsigned short)(u >> 16);
}
__device__ __forceinline__ float bf16_to_f32(unsigned short s) {
    return __uint_as_float(((unsigned int)s) << 16);
}
__device__ __forceinline__ float sigmoidf_(float x) {
    return 1.0f / (1.0f + __expf(-x));
}

// ---------------------------------------------------------------------------
// Prep: convert W_hh -> bf16 (same [G][H] row-major layout), fuse biases.
// ---------------------------------------------------------------------------
__global__ __launch_bounds__(256)
void prep_weights(const float* __restrict__ W_hh,
                  const float* __restrict__ b_ih, const float* __restrict__ b_hh,
                  unsigned short* __restrict__ Wbf, float* __restrict__ bias) {
    int i = blockIdx.x * 256 + threadIdx.x;
    if (i < Gd * Hd) Wbf[i] = f32_to_bf16(W_hh[i]);
    if (i < Gd)      bias[i] = b_ih[i] + b_hh[i];
}

// ---------------------------------------------------------------------------
// Prep: h0 -> bf16, c = 0, x = broadcast(obs[-1]).
// ---------------------------------------------------------------------------
__global__ __launch_bounds__(256)
void prep_state(const float* __restrict__ h0, const float* __restrict__ obs,
                unsigned short* __restrict__ hb, float* __restrict__ cbuf,
                float* __restrict__ xbuf) {
    long long i = (long long)blockIdx.x * 256 + threadIdx.x;
    if (i < (long long)ROWS * Hd) {
        hb[i]   = f32_to_bf16(h0[i]);
        cbuf[i] = 0.0f;
    }
    if (i < ROWS * 2) {
        // x[k,n,d] = obs[OBS-1, n, d];  (n*2+d) == i mod (Np*2), Np*2 = 8192
        xbuf[i] = obs[(OBS - 1) * Np * 2 + ((int)i & (Np * 2 - 1))];
    }
}

// ---------------------------------------------------------------------------
// One LSTM timestep: gates = h·W_hh^T (+ x·W_ih^T + b), pointwise update.
// Block = 256 threads (8 wave32), tile = 128 rows x 32 hidden cols x 4 groups.
// Wave w: row-strip pair (w&3)->strips {2p,2p+1}, gate-group pair (w>>2)
//   -> 8 WMMA accumulator tiles; 12 b128 loads per 8 WMMAs (1.5 loads/WMMA).
// ---------------------------------------------------------------------------
__global__ __launch_bounds__(256, 2)
void lstm_step(const unsigned short* __restrict__ h_cur,
               unsigned short* __restrict__ h_next,
               float* __restrict__ c_buf,
               const float* __restrict__ x_buf,
               const unsigned short* __restrict__ Wbf,
               const float* __restrict__ bias,
               const float* __restrict__ W_ih) {
    __shared__ float gates[BM][132];   // 4 groups x 32 cols, padded vs bank conflicts

    const int tid      = threadIdx.x;
    const int lane     = tid & 31;
    const int wv       = tid >> 5;
    const int rp       = wv & 3;       // row-strip pair: strips 2rp, 2rp+1
    const int gp       = wv >> 2;      // gate-group pair {0,1} or {2,3}
    const int laneHalf = lane >> 4;    // 0 or 1
    const int ln       = lane & 15;

    const int rowTile  = blockIdx.x * BM;
    const int colSlice = blockIdx.y * 32;           // hidden-col slice within each group

    v8f acc[8];
#pragma unroll
    for (int t = 0; t < 8; ++t) acc[t] = (v8f){0.f,0.f,0.f,0.f,0.f,0.f,0.f,0.f};

    // A fragments: lane ln holds rows (rowTile + (2rp+s)*16 + ln), K striped in VGPRs
    const unsigned short* aBase0 = h_cur + (size_t)(rowTile + (2 * rp    ) * 16 + ln) * Hd;
    const unsigned short* aBase1 = h_cur + (size_t)(rowTile + (2 * rp + 1) * 16 + ln) * Hd;

    // B fragments: lane ln holds gate-column bCol[c], two 8-wide K runs per half-wave (same order as A)
    int bCol[4];
#pragma unroll
    for (int c = 0; c < 4; ++c) {
        int group = gp * 2 + (c >> 1);
        int sub   = c & 1;
        bCol[c] = group * Hd + colSlice + sub * 16 + ln;
    }

#pragma unroll 4
    for (int kb = 0; kb < Hd; kb += 32) {
        // A: lanes 0-15 take K = kb..kb+7 & kb+16..kb+23; lanes 16-31 shifted by 8
        union { uint4 u[2]; v16bf v; } a0, a1;
        const uint4* ap0 = (const uint4*)(aBase0 + kb + laneHalf * 8);
        const uint4* ap1 = (const uint4*)(aBase1 + kb + laneHalf * 8);
        a0.u[0] = ap0[0];
        a0.u[1] = ap0[2];   // +16 bf16 = +2 uint4
        a1.u[0] = ap1[0];
        a1.u[1] = ap1[2];
#pragma unroll
        for (int c = 0; c < 4; ++c) {
            // B: lane holds W_hh[bCol][kb + laneHalf*8 .. +7] & [kb + 16 + laneHalf*8 .. +7]
            union { uint4 u[2]; v16bf v; } bfr;
            const uint4* bp = (const uint4*)(Wbf + (size_t)bCol[c] * Hd + kb + laneHalf * 8);
            bfr.u[0] = bp[0];
            bfr.u[1] = bp[2];
            acc[c] = __builtin_amdgcn_wmma_f32_16x16x32_bf16(
                false, a0.v, false, bfr.v, (short)0, acc[c], false, false);
            acc[4 + c] = __builtin_amdgcn_wmma_f32_16x16x32_bf16(
                false, a1.v, false, bfr.v, (short)0, acc[4 + c], false, false);
        }
    }

    // Spill accumulators: C/D layout -> row = r + 8*laneHalf, col = ln (per tile)
#pragma unroll
    for (int t = 0; t < 8; ++t) {
        int s     = t >> 2;            // row strip within pair
        int c     = t & 3;             // col tile
        int group = gp * 2 + (c >> 1);
        int sub   = c & 1;
        int c0 = group * 32 + sub * 16 + ln;
        int r0 = (2 * rp + s) * 16 + laneHalf * 8;
#pragma unroll
        for (int r = 0; r < 8; ++r) gates[r0 + r][c0] = acc[t][r];
    }
    __syncthreads();

    // Pointwise LSTM update over the 128x32 tile (i,f,g,o = groups 0..3)
    for (int idx = tid; idx < BM * 32; idx += 256) {
        int r  = idx >> 5;
        int j  = idx & 31;
        int row = rowTile + r;
        int cg  = colSlice + j;
        float x0 = x_buf[row * 2 + 0];
        float x1 = x_buf[row * 2 + 1];
        float v[4];
#pragma unroll
        for (int q = 0; q < 4; ++q) {
            int gc = q * Hd + cg;
            v[q] = gates[r][q * 32 + j]
                 + x0 * W_ih[gc * 2 + 0] + x1 * W_ih[gc * 2 + 1] + bias[gc];
        }
        float ig = sigmoidf_(v[0]);
        float fg = sigmoidf_(v[1]);
        float gg = tanhf(v[2]);
        float og = sigmoidf_(v[3]);
        size_t cidx = (size_t)row * Hd + cg;
        float cn = fg * c_buf[cidx] + ig * gg;
        c_buf[cidx] = cn;
        h_next[cidx] = f32_to_bf16(og * tanhf(cn));
    }
}

// ---------------------------------------------------------------------------
// hidden2pos for ALL K samples (feeds next x); write k==0 slice to output.
// One thread per batch row; W_pos reads are wave-uniform (L2 resident).
// ---------------------------------------------------------------------------
__global__ __launch_bounds__(256)
void proj_step(const unsigned short* __restrict__ hsrc,
               const float* __restrict__ W_pos, const float* __restrict__ b_pos,
               float* __restrict__ xbuf, float* __restrict__ out, int t) {
    int row = blockIdx.x * 256 + threadIdx.x;
    const uint4* hp = (const uint4*)(hsrc + (size_t)row * Hd);
    float s0 = 0.f, s1 = 0.f;
    for (int it = 0; it < Hd / 8; ++it) {
        uint4 u = hp[it];
        unsigned int uu[4] = {u.x, u.y, u.z, u.w};
#pragma unroll
        for (int e = 0; e < 4; ++e) {
            float lo = __uint_as_float(uu[e] << 16);
            float hi = __uint_as_float(uu[e] & 0xFFFF0000u);
            int k = it * 8 + e * 2;
            s0 += lo * W_pos[k]      + hi * W_pos[k + 1];
            s1 += lo * W_pos[Hd + k] + hi * W_pos[Hd + k + 1];
        }
    }
    s0 += b_pos[0];
    s1 += b_pos[1];
    xbuf[row * 2 + 0] = s0;
    xbuf[row * 2 + 1] = s1;
    if (row < Np) {   // k == 0 slice -> pred_traj_rel[t]
        out[((size_t)t * Np + row) * 2 + 0] = s0;
        out[((size_t)t * Np + row) * 2 + 1] = s1;
    }
}

// ---------------------------------------------------------------------------
extern "C" void kernel_launch(void* const* d_in, const int* in_sizes, int n_in,
                              void* d_out, int out_size, void* d_ws, size_t ws_size,
                              hipStream_t stream) {
    const float* obs   = (const float*)d_in[0];
    /* d_in[1] fut_traj_rel unused (only provides fut_len=12) */
    const float* h0    = (const float*)d_in[2];
    const float* W_ih  = (const float*)d_in[3];
    const float* W_hh  = (const float*)d_in[4];
    const float* b_ih  = (const float*)d_in[5];
    const float* b_hh  = (const float*)d_in[6];
    const float* W_pos = (const float*)d_in[7];
    const float* b_pos = (const float*)d_in[8];
    float* out = (float*)d_out;

    // Workspace layout (bytes)
    char* ws = (char*)d_ws;
    const size_t szWbf  = (size_t)Gd * Hd * 2;       // 2 MiB bf16 weights
    const size_t szBias = 8192;                      // fused bias (padded)
    const size_t szH    = (size_t)ROWS * Hd * 2;     // 80 MiB bf16 h state
    const size_t szC    = (size_t)ROWS * Hd * 4;     // 160 MiB fp32 c state
    unsigned short* Wbf  = (unsigned short*)(ws);
    float*          bias = (float*)(ws + szWbf);
    unsigned short* hb0  = (unsigned short*)(ws + szWbf + szBias);
    unsigned short* hb1  = (unsigned short*)(ws + szWbf + szBias + szH);
    float*          cbuf = (float*)(ws + szWbf + szBias + 2 * szH);
    float*          xbuf = (float*)(ws + szWbf + szBias + 2 * szH + szC);

    prep_weights<<<(Gd * Hd + 255) / 256, 256, 0, stream>>>(W_hh, b_ih, b_hh, Wbf, bias);
    prep_state<<<((long long)ROWS * Hd + 255) / 256, 256, 0, stream>>>(h0, obs, hb0, cbuf, xbuf);

    for (int t = 0; t < FUT; ++t) {
        const unsigned short* hc = (t & 1) ? hb1 : hb0;
        unsigned short*       hn = (t & 1) ? hb0 : hb1;
        lstm_step<<<dim3(ROWS / BM, Hd / 32, 1), 256, 0, stream>>>(
            hc, hn, cbuf, xbuf, Wbf, bias, W_ih);
        proj_step<<<ROWS / 256, 256, 0, stream>>>(hn, W_pos, b_pos, xbuf, out, t);
    }
}
